// FGS1MambaEncoder_57037165691705
// MI455X (gfx1250) — compile-verified, matrix operand order checked by reading, not run on hardware
//
#include <hip/hip_runtime.h>
#include <hip/hip_bf16.h>
#include <math.h>

typedef _Float16 v16h __attribute__((ext_vector_type(16)));
typedef _Float16 v8h  __attribute__((ext_vector_type(8)));
typedef float    v8f  __attribute__((ext_vector_type(8)));
typedef int      v4i  __attribute__((ext_vector_type(4)));

static constexpr int kB   = 2;
static constexpr int kLin = 135000;
static constexpr int kT   = 4218;          // (135000-32)/32 + 1
static constexpr int kN   = kB * kT;       // 8436 tokens
static constexpr int kD   = 768;
static constexpr int kE   = 1536;
static constexpr int kS   = 16;
static constexpr int kR   = 48;
static constexpr int kLayers = 24;

static constexpr int BM = 128;
static constexpr int BN = 64;
static constexpr int BK = 32;
static constexpr int LDT = 40;             // padded LDS row stride (halfs); 80B, 16B-aligned

// chunked scan: 32 chunks per batch, 132 steps each (last chunk 126)
static constexpr int kC = 32;
static constexpr int kL = 132;

// ---------------- CDNA5 async global->LDS helpers (guarded) ----------------
#if __has_builtin(__builtin_amdgcn_global_load_async_to_lds_b128)
#define ASYNC_A_TILE 1
typedef __attribute__((address_space(1))) v4i* g4p;   // global 16B chunk
typedef __attribute__((address_space(3))) v4i* l4p;   // LDS 16B chunk
__device__ __forceinline__ void async_cp16(const _Float16* g, _Float16* l)
{
  __builtin_amdgcn_global_load_async_to_lds_b128((g4p)(void*)g, (l4p)(void*)l, 0, 0);
}
__device__ __forceinline__ void wait_async0()
{
#if __has_builtin(__builtin_amdgcn_s_wait_asynccnt)
  __builtin_amdgcn_s_wait_asynccnt(0);
#else
  asm volatile("s_wait_asynccnt 0" ::: "memory");
#endif
}
#endif

// ---------------- downsample conv: x (B,1,135000) -> h (N, 768) ----------------
__global__ void ds_conv_kernel(const float* __restrict__ x, const float* __restrict__ cw,
                               const float* __restrict__ cb, float* __restrict__ h)
{
  __shared__ float win[32];
  int tok = blockIdx.x;
  int b = tok / kT, t = tok % kT;
  if (threadIdx.x < 32) win[threadIdx.x] = x[(size_t)b * kLin + (size_t)t * 32 + threadIdx.x];
  __syncthreads();
  for (int d = threadIdx.x; d < kD; d += 256) {
    const float* wd = cw + d * 32;
    float acc = cb[d];
#pragma unroll
    for (int k = 0; k < 32; ++k) acc += win[k] * wd[k];
    h[(size_t)tok * kD + d] = acc;
  }
}

// ---------------- RMSNorm per token (768), output f16 or f32 ----------------
template<int OUT_HALF>
__global__ void rmsnorm_kernel(const float* __restrict__ h, const float* __restrict__ w,
                               void* __restrict__ outp)
{
  __shared__ float red[256];
  int row = blockIdx.x;
  const float* xr = h + (size_t)row * kD;
  float ss = 0.f;
  for (int i = threadIdx.x; i < kD; i += 256) { float v = xr[i]; ss += v * v; }
  red[threadIdx.x] = ss;
  __syncthreads();
  for (int o = 128; o > 0; o >>= 1) {
    if (threadIdx.x < o) red[threadIdx.x] += red[threadIdx.x + o];
    __syncthreads();
  }
  float inv = rsqrtf(red[0] / (float)kD + 1e-5f);
  for (int i = threadIdx.x; i < kD; i += 256) {
    float v = xr[i] * inv * w[i];
    if (OUT_HALF) ((_Float16*)outp)[(size_t)row * kD + i] = (_Float16)v;
    else          ((float*)outp)[(size_t)row * kD + i] = v;
  }
}

// ---------------- WMMA f16 GEMM: C[M,Nd] = A_f16[M,K] * W_f32[Nd,K]^T ----------------
// EPI: 0 = store, 1 = add into C (residual), 2 = softplus(acc + bias[col])
// NOTE: all launched K are multiples of 16, so 8-half / 4-float chunks are never
// partially in-range -> whole-chunk zero-fill, no scalar fallback.
__device__ __forceinline__ v16h frag_ld(const _Float16* base, int lane)
{
  int m  = lane & 15;
  int kq = (lane >> 4) << 3;               // 0 or 8
  const _Float16* p = base + m * LDT + kq;
  union { v16h v; v8h h[2]; } r;
  r.h[0] = *(const v8h*)p;                 // K = kq .. kq+7
  r.h[1] = *(const v8h*)(p + 16);          // K = kq+16 .. kq+23
  return r.v;
}

template<int EPI>
__global__ void __launch_bounds__(256) gemm_f16_kernel(
    const _Float16* __restrict__ A, const float* __restrict__ W,
    float* __restrict__ C, const float* __restrict__ bias,
    int M, int Nd, int K)
{
  __shared__ _Float16 As[BM * LDT];
  __shared__ _Float16 Bs[BN * LDT];
  int tid  = threadIdx.x;
  int lane = tid & 31;
  int wid  = tid >> 5;
  int wrow = wid & 3;                       // 4 waves over M (32 rows each)
  int wcol = wid >> 2;                      // 2 waves over N (32 cols each)
  int blockM = blockIdx.x * BM;
  int blockN = blockIdx.y * BN;
#ifdef ASYNC_A_TILE
  const bool asyncA = ((blockM + BM) <= M) && ((K & (BK - 1)) == 0);  // block-uniform
#endif

  v8f acc[2][2];
#pragma unroll
  for (int i = 0; i < 2; ++i)
#pragma unroll
    for (int j = 0; j < 2; ++j)
#pragma unroll
      for (int e = 0; e < 8; ++e) acc[i][j][e] = 0.f;

  for (int k0 = 0; k0 < K; k0 += BK) {
    // ---- stage A tile ----
#ifdef ASYNC_A_TILE
    if (asyncA) {
#pragma unroll
      for (int jj = 0; jj < 2; ++jj) {
        int cch = tid + jj * 256;           // 512 chunks of 8 halfs (16B)
        int r   = cch >> 2;
        int kc  = (cch & 3) << 3;
        async_cp16(A + (size_t)(blockM + r) * K + k0 + kc, &As[r * LDT + kc]);
      }
    } else
#endif
    {
#pragma unroll
      for (int jj = 0; jj < 2; ++jj) {
        int cch = tid + jj * 256;
        int r   = cch >> 2;
        int kc  = (cch & 3) << 3;
        int g   = blockM + r;
        v8h val;
        if (g < M && (k0 + kc + 8) <= K) {
          val = *(const v8h*)(A + (size_t)g * K + k0 + kc);
        } else {
#pragma unroll
          for (int e = 0; e < 8; ++e) val[e] = (_Float16)0.f;
        }
        *(v8h*)&As[r * LDT + kc] = val;
      }
    }
    // ---- stage W tile, f32 -> f16 on the fly ----
#pragma unroll
    for (int jj = 0; jj < 2; ++jj) {
      int cch = tid + jj * 256;             // 512 chunks of 4 floats
      int r   = cch >> 3;
      int kc  = (cch & 7) << 2;
      int n   = blockN + r;
      _Float16* dst = &Bs[r * LDT + kc];
      if (n < Nd && (k0 + kc + 4) <= K) {
        float4 w4 = *(const float4*)(W + (size_t)n * K + k0 + kc);
        dst[0] = (_Float16)w4.x; dst[1] = (_Float16)w4.y;
        dst[2] = (_Float16)w4.z; dst[3] = (_Float16)w4.w;
      } else {
        dst[0] = (_Float16)0.f; dst[1] = (_Float16)0.f;
        dst[2] = (_Float16)0.f; dst[3] = (_Float16)0.f;
      }
    }
#ifdef ASYNC_A_TILE
    if (asyncA) wait_async0();              // A-tile DMA complete before barrier
#endif
    __syncthreads();

    v16h a0 = frag_ld(&As[(wrow * 32 +  0) * LDT], lane);
    v16h a1 = frag_ld(&As[(wrow * 32 + 16) * LDT], lane);
    v16h b0 = frag_ld(&Bs[(wcol * 32 +  0) * LDT], lane);
    v16h b1 = frag_ld(&Bs[(wcol * 32 + 16) * LDT], lane);
    acc[0][0] = __builtin_amdgcn_wmma_f32_16x16x32_f16(false, a0, false, b0, (short)0, acc[0][0], false, false);
    acc[0][1] = __builtin_amdgcn_wmma_f32_16x16x32_f16(false, a0, false, b1, (short)0, acc[0][1], false, false);
    acc[1][0] = __builtin_amdgcn_wmma_f32_16x16x32_f16(false, a1, false, b0, (short)0, acc[1][0], false, false);
    acc[1][1] = __builtin_amdgcn_wmma_f32_16x16x32_f16(false, a1, false, b1, (short)0, acc[1][1], false, false);
    __syncthreads();
  }

  // epilogue: C/D layout — VGPR r: lanes 0-15 -> M=r, lanes 16-31 -> M=r+8; N = lane&15
  int n_off = lane & 15;
  int r_add = (lane >> 4) << 3;
#pragma unroll
  for (int i = 0; i < 2; ++i) {
#pragma unroll
    for (int j = 0; j < 2; ++j) {
      int rowT = blockM + wrow * 32 + i * 16;
      int colT = blockN + wcol * 32 + j * 16;
#pragma unroll
      for (int r = 0; r < 8; ++r) {
        int row = rowT + r + r_add;
        int col = colT + n_off;
        if (row < M && col < Nd) {
          float v = acc[i][j][r];
          size_t idx = (size_t)row * Nd + col;
          if (EPI == 0)      C[idx] = v;
          else if (EPI == 1) C[idx] += v;
          else {
            float z = v + bias[col];
            C[idx] = (z > 20.f) ? z : log1pf(__expf(z));
          }
        }
      }
    }
  }
}

// ---------------- causal depthwise conv (K=4) + SiLU ----------------
__global__ void dwconv_silu_kernel(const float* __restrict__ xz, const float* __restrict__ cw,
                                   const float* __restrict__ cb,
                                   float* __restrict__ u32, _Float16* __restrict__ u16)
{
  long idx = (long)blockIdx.x * blockDim.x + threadIdx.x;   // over N*E
  if (idx >= (long)kN * kE) return;
  int ch  = (int)(idx % kE);
  int tok = (int)(idx / kE);
  int b = tok / kT, t = tok % kT;
  const float* w = cw + ch * 4;
  float acc = cb[ch];
#pragma unroll
  for (int j = 0; j < 4; ++j) {
    int tt = t - 3 + j;
    if (tt >= 0) acc += xz[(size_t)(b * kT + tt) * (2 * kE) + ch] * w[j];
  }
  float sv = acc / (1.f + __expf(-acc));                    // SiLU
  u32[idx] = sv;
  u16[idx] = (_Float16)sv;
}

// ---------------- split dt-rank columns of proj into f16 GEMM input ----------------
__global__ void split_dt_kernel(const float* __restrict__ proj, _Float16* __restrict__ dt16)
{
  long idx = (long)blockIdx.x * blockDim.x + threadIdx.x;   // over N*R
  if (idx >= (long)kN * kR) return;
  int row = (int)(idx / kR), j = (int)(idx % kR);
  dt16[idx] = (_Float16)proj[(size_t)row * 80 + j];
}

// ---------------- chunked selective scan ----------------
// Linear recurrence s_t = a_t*s + b_t with a_t[n] = exp(dt*Aneg[n]); over a chunk the
// product of a_t collapses to exp(Aneg[n]*sum(dt)), so chunk stitching needs only sum(dt).
// pass1: per-chunk local scan from zero init -> slocal, sumdt   (kB*kC*kE threads)
// pass2: sequential stitch of kC chunk states per channel -> sinit  (kB*kE threads)
// pass3: re-scan each chunk from sinit, fused D-skip + SiLU(z) gate -> y16 (f16)
__global__ void scan_pass1(const float* __restrict__ delta, const float* __restrict__ u,
                           const float* __restrict__ proj, const float* __restrict__ A_log,
                           float* __restrict__ slocal, float* __restrict__ sumdt)
{
  int gidx = blockIdx.x * blockDim.x + threadIdx.x;         // (b, c, ch), ch fastest
  if (gidx >= kB * kC * kE) return;
  int ch = gidx % kE;
  int bc = gidx / kE;                                       // b*kC + c
  int c  = bc % kC;
  int b  = bc / kC;
  int t0 = c * kL;
  int t1 = (t0 + kL < kT) ? (t0 + kL) : kT;

  float Aneg[kS];
#pragma unroll
  for (int n = 0; n < kS; ++n) Aneg[n] = -__expf(A_log[ch * kS + n]);
  float s[kS];
#pragma unroll
  for (int n = 0; n < kS; ++n) s[n] = 0.f;
  float sd = 0.f;

  for (int t = t0; t < t1; ++t) {
    size_t row = (size_t)b * kT + t;
    if (t + 8 < t1) {
      __builtin_prefetch((const void*)(delta + (row + 8) * kE + ch), 0, 1);
      __builtin_prefetch((const void*)(u + (row + 8) * kE + ch), 0, 1);
    }
    float dt = delta[row * kE + ch];
    float ut = u[row * kE + ch];
    const float* pr = proj + row * 80;                      // cols 48..63 = B
    float du = dt * ut;
    sd += dt;
#pragma unroll
    for (int n = 0; n < kS; ++n)
      s[n] = __expf(dt * Aneg[n]) * s[n] + du * pr[48 + n];
  }
#pragma unroll
  for (int n = 0; n < kS; ++n) slocal[((size_t)bc * kS + n) * kE + ch] = s[n];
  sumdt[(size_t)bc * kE + ch] = sd;
}

__global__ void scan_pass2(const float* __restrict__ A_log, const float* __restrict__ slocal,
                           const float* __restrict__ sumdt, float* __restrict__ sinit)
{
  int idx = blockIdx.x * blockDim.x + threadIdx.x;          // over kB*kE
  if (idx >= kB * kE) return;
  int ch = idx % kE, b = idx / kE;
  float Aneg[kS];
#pragma unroll
  for (int n = 0; n < kS; ++n) Aneg[n] = -__expf(A_log[ch * kS + n]);
  float s[kS];
#pragma unroll
  for (int n = 0; n < kS; ++n) s[n] = 0.f;
  for (int c = 0; c < kC; ++c) {
    size_t bc = (size_t)b * kC + c;
#pragma unroll
    for (int n = 0; n < kS; ++n) sinit[(bc * kS + n) * kE + ch] = s[n];
    float sd = sumdt[bc * kE + ch];
#pragma unroll
    for (int n = 0; n < kS; ++n)
      s[n] = __expf(Aneg[n] * sd) * s[n] + slocal[(bc * kS + n) * kE + ch];
  }
}

__global__ void scan_pass3(const float* __restrict__ delta, const float* __restrict__ u,
                           const float* __restrict__ proj, const float* __restrict__ xz,
                           const float* __restrict__ A_log, const float* __restrict__ Dskip,
                           const float* __restrict__ sinit, _Float16* __restrict__ y16)
{
  int gidx = blockIdx.x * blockDim.x + threadIdx.x;         // (b, c, ch), ch fastest
  if (gidx >= kB * kC * kE) return;
  int ch = gidx % kE;
  int bc = gidx / kE;
  int c  = bc % kC;
  int b  = bc / kC;
  int t0 = c * kL;
  int t1 = (t0 + kL < kT) ? (t0 + kL) : kT;

  float Aneg[kS];
#pragma unroll
  for (int n = 0; n < kS; ++n) Aneg[n] = -__expf(A_log[ch * kS + n]);
  float Dv = Dskip[ch];
  float s[kS];
#pragma unroll
  for (int n = 0; n < kS; ++n) s[n] = sinit[((size_t)bc * kS + n) * kE + ch];

  for (int t = t0; t < t1; ++t) {
    size_t row = (size_t)b * kT + t;
    if (t + 8 < t1) {
      __builtin_prefetch((const void*)(delta + (row + 8) * kE + ch), 0, 1);
      __builtin_prefetch((const void*)(u + (row + 8) * kE + ch), 0, 1);
      __builtin_prefetch((const void*)(xz + (row + 8) * (2 * kE) + kE + ch), 0, 1);
    }
    float dt = delta[row * kE + ch];
    float ut = u[row * kE + ch];
    const float* pr = proj + row * 80;                      // B: 48..63, C: 64..79
    float du = dt * ut;
    float y = 0.f;
#pragma unroll
    for (int n = 0; n < kS; ++n) {
      s[n] = __expf(dt * Aneg[n]) * s[n] + du * pr[48 + n];
      y += s[n] * pr[64 + n];
    }
    float zt = xz[row * (2 * kE) + kE + ch];
    float g = zt / (1.f + __expf(-zt));                     // SiLU(z)
    y16[row * kE + ch] = (_Float16)((y + ut * Dv) * g);
  }
}

// ---------------- mean pool over time ----------------
__global__ void pool_kernel(const float* __restrict__ normed, float* __restrict__ pooled)
{
  int d = blockIdx.x * blockDim.x + threadIdx.x;            // over B*D
  if (d >= kB * kD) return;
  int b = d / kD, dd = d % kD;
  float sum = 0.f;
  for (int t = 0; t < kT; ++t) sum += normed[(size_t)(b * kT + t) * kD + dd];
  pooled[d] = sum / (float)kT;
}

// ---------------- head: 768->64 projection + LayerNorm(64) ----------------
__global__ void head_kernel(const float* __restrict__ pooled, const float* __restrict__ pw,
                            const float* __restrict__ pb, const float* __restrict__ lnw,
                            const float* __restrict__ lnb, float* __restrict__ out)
{
  __shared__ float zed[kB * 64];
  int tid = threadIdx.x;                                    // 0..127
  int b = tid >> 6, j = tid & 63;
  float acc = pb[j];
  const float* pr = pooled + b * kD;
  const float* wr = pw + j * kD;
  for (int k = 0; k < kD; ++k) acc += pr[k] * wr[k];
  zed[tid] = acc;
  __syncthreads();
  float mu = 0.f;
  for (int k = 0; k < 64; ++k) mu += zed[b * 64 + k];
  mu *= (1.f / 64.f);
  float var = 0.f;
  for (int k = 0; k < 64; ++k) { float d = zed[b * 64 + k] - mu; var += d * d; }
  var *= (1.f / 64.f);
  out[tid] = (acc - mu) * rsqrtf(var + 1e-5f) * lnw[j] + lnb[j];
}

extern "C" void kernel_launch(void* const* d_in, const int* in_sizes, int n_in,
                              void* d_out, int out_size, void* d_ws, size_t ws_size,
                              hipStream_t stream)
{
  const float* x       = (const float*)d_in[0];
  const float* conv_w  = (const float*)d_in[1];
  const float* conv_b  = (const float*)d_in[2];
  const float* norm_w  = (const float*)d_in[3];
  const float* in_proj = (const float*)d_in[4];
  const float* c1_w    = (const float*)d_in[5];
  const float* c1_b    = (const float*)d_in[6];
  const float* xp_w    = (const float*)d_in[7];
  const float* dt_w    = (const float*)d_in[8];
  const float* dt_b    = (const float*)d_in[9];
  const float* A_log   = (const float*)d_in[10];
  const float* Dskip   = (const float*)d_in[11];
  const float* op_w    = (const float*)d_in[12];
  const float* norm_f  = (const float*)d_in[13];
  const float* proj_w  = (const float*)d_in[14];
  const float* proj_b  = (const float*)d_in[15];
  const float* ln_w    = (const float*)d_in[16];
  const float* ln_b    = (const float*)d_in[17];
  float* out = (float*)d_out;

  char* p = (char*)d_ws;
  auto alloc = [&](size_t bytes) -> void* {
    void* r = (void*)p;
    p += (bytes + 255) & ~(size_t)255;
    return r;
  };
  float*    h      = (float*)   alloc((size_t)kN * kD * 4);        // residual stream
  float*    xz     = (float*)   alloc((size_t)kN * 2 * kE * 4);    // in_proj out; reused as f32 norm scratch
  _Float16* nrm16  = (_Float16*)alloc((size_t)kN * kD * 2);
  float*    u32    = (float*)   alloc((size_t)kN * kE * 4);
  _Float16* u16    = (_Float16*)alloc((size_t)kN * kE * 2);
  float*    projb  = (float*)   alloc((size_t)kN * 80 * 4);        // dt(48) | B(16) | C(16)
  _Float16* dt16   = (_Float16*)alloc((size_t)kN * kR * 2);
  float*    delta  = (float*)   alloc((size_t)kN * kE * 4);
  _Float16* y16    = (_Float16*)alloc((size_t)kN * kE * 2);
  float*    slocal = (float*)   alloc((size_t)kB * kC * kS * kE * 4);
  float*    sinit  = (float*)   alloc((size_t)kB * kC * kS * kE * 4);
  float*    sumdt  = (float*)   alloc((size_t)kB * kC * kE * 4);
  float*    pooled = (float*)   alloc((size_t)kB * kD * 4);

  // downsample conv -> h
  ds_conv_kernel<<<kN, 256, 0, stream>>>(x, conv_w, conv_b, h);

  const int MB = (kN + BM - 1) / BM;   // 66 M-tiles
  const int nScan1 = (kB * kC * kE + 255) / 256;   // 384 blocks

  for (int l = 0; l < kLayers; ++l) {
    rmsnorm_kernel<1><<<kN, 256, 0, stream>>>(h, norm_w + (size_t)l * kD, nrm16);

    gemm_f16_kernel<0><<<dim3(MB, (2 * kE) / BN), 256, 0, stream>>>(
        nrm16, in_proj + (size_t)l * 2 * kE * kD, xz, nullptr, kN, 2 * kE, kD);

    {
      long tot = (long)kN * kE;
      dwconv_silu_kernel<<<(int)((tot + 255) / 256), 256, 0, stream>>>(
          xz, c1_w + (size_t)l * kE * 4, c1_b + (size_t)l * kE, u32, u16);
    }

    gemm_f16_kernel<0><<<dim3(MB, (80 + BN - 1) / BN), 256, 0, stream>>>(
        u16, xp_w + (size_t)l * 80 * kE, projb, nullptr, kN, 80, kE);

    {
      long tot = (long)kN * kR;
      split_dt_kernel<<<(int)((tot + 255) / 256), 256, 0, stream>>>(projb, dt16);
    }

    gemm_f16_kernel<2><<<dim3(MB, kE / BN), 256, 0, stream>>>(
        dt16, dt_w + (size_t)l * kE * kR, delta, dt_b + (size_t)l * kE, kN, kE, kR);

    const float* Al = A_log + (size_t)l * kE * kS;
    scan_pass1<<<nScan1, 256, 0, stream>>>(delta, u32, projb, Al, slocal, sumdt);
    scan_pass2<<<(kB * kE + 255) / 256, 256, 0, stream>>>(Al, slocal, sumdt, sinit);
    scan_pass3<<<nScan1, 256, 0, stream>>>(delta, u32, projb, xz, Al,
                                           Dskip + (size_t)l * kE, sinit, y16);

    gemm_f16_kernel<1><<<dim3(MB, kD / BN), 256, 0, stream>>>(
        y16, op_w + (size_t)l * kD * kE, h, nullptr, kN, kD, kE);
  }

  // final RMSNorm (f32 into reused xz scratch) -> mean pool -> head
  rmsnorm_kernel<0><<<kN, 256, 0, stream>>>(h, norm_f, xz);
  pool_kernel<<<(kB * kD + 255) / 256, 256, 0, stream>>>(xz, pooled);
  head_kernel<<<1, 128, 0, stream>>>(pooled, proj_w, proj_b, ln_w, ln_b, out);

  (void)in_sizes; (void)n_in; (void)out_size; (void)ws_size;
}
